// VqVAE_58866821759049
// MI455X (gfx1250) — matrix operand order repaired and checked by reading, not compiled, not run on hardware
//
#include <hip/hip_runtime.h>
#include <hip/hip_bf16.h>

// ---------------------------------------------------------------------------
// VQ-VAE fused forward for MI455X (gfx1250, wave32, WMMA bf16).
//
// Roofline: ~704 MB mandatory HBM traffic (570 MB f32 weights read once,
// 67 MB x, 67 MB out) vs 73 GFLOP => memory bound at 23.3 TB/s (~30us).
// Strategy: fuse the entire 6-GEMM + VQ chain per feature f into one kernel;
// all intermediates live in the 320KB-per-WGP LDS; math in
// v_wmma_f32_16x16x32_bf16 (f32 accumulate) so compute never bottlenecks;
// weight slabs are software-pipelined (global loads for slab k+1 in flight
// while slab k's WMMAs run) with ping-pong LDS staging and ONE barrier/slab.
// ---------------------------------------------------------------------------

#define FTS     256
#define IN_DIM  255
#define U0      512
#define U1      256
#define DIMQ    64
#define EMB     512
#define BATCH   256

#define MTILE   128        // batch rows per block (2 blocks per feature)
#define NTHREADS 256       // 8 wave32s
#define WT_PITCH 40        // padded K-pitch of staged weight tile (bank-safe)
#define APAD     8         // activation pitch pad (bank-safe LDS reads)
#define WT_BUFELEMS (128 * WT_PITCH)   // one ping-pong weight buffer (ushorts)

typedef __attribute__((ext_vector_type(16))) __bf16 v16bf;
typedef __attribute__((ext_vector_type(8)))  float  v8f;

union BF16x16 { unsigned int u[8]; v16bf v; };

__device__ __forceinline__ unsigned short f2bf(float x) {
    unsigned int u = __float_as_uint(x);
    unsigned int r = u + 0x7FFFu + ((u >> 16) & 1u);   // round-nearest-even
    return (unsigned short)(r >> 16);
}

// ---------------------------------------------------------------------------
// One tiled GEMM stage: D[128 x N] = act( A[128 x K](bf16, LDS) * W[K x N] + b )
//   MODE 0: relu -> O (LDS, bf16)
//   MODE 1: sigmoid -> gout (global f32, out[b][f][n] layout)
//   MODE 2: VQ scoring: s = |c_n|^2 - 2*acc  -> running per-row argmin
// Weight K-slabs (32 x NC) stream from HBM software-pipelined: loads for
// slab kt+1 are issued before the WMMA burst of slab kt; f32->bf16 convert +
// ds_store into the ping-pong buffer after the WMMAs; one barrier per slab.
// ---------------------------------------------------------------------------
template <int NC, int MODE, int WTR>
__device__ __forceinline__ void fat_gemm(
    const float* __restrict__ W, int K, int N, int ldw,
    const unsigned short* __restrict__ A, int apitch,
    unsigned short* __restrict__ O, int opitch,
    const float* __restrict__ bias,
    float* __restrict__ gout, int f, int b0,
    unsigned short* __restrict__ Wt,
    float* __restrict__ Sc, const float* __restrict__ cnorm,
    float* __restrict__ redv, int* __restrict__ redi,
    float* __restrict__ minv, int* __restrict__ mini)
{
    const int lane  = threadIdx.x & 31;
    const int wave  = threadIdx.x >> 5;
    const int m0    = wave * 16;
    const int colin = lane & 15;
    const int halfsel = (lane < 16) ? 0 : 16;   // A-frag K base select
    const int ksel    = (lane < 16) ? 0 : 16;   // B-frag K base select
    const int rofs    = (lane < 16) ? 0 : 8;    // C/D row offset

    constexpr int NTILES = NC / 16;             // accumulator tiles per wave
    constexpr int LEL    = (NC * 32) / NTHREADS;// staged elems per thread
    const int nchunks = N / NC;
    const int ktiles  = ((K + 31) & ~31) >> 5;

    for (int nc = 0; nc < nchunks; ++nc) {
        const int n0 = nc * NC;
        v8f acc[NTILES] = {};
        float rw[LEL];

        // ---- prologue: fetch + stage slab 0 into Wt buffer 0 ----
#pragma unroll
        for (int i = 0; i < LEL; ++i) {
            int e  = i * NTHREADS + threadIdx.x;
            int nn = e % NC, kk = e / NC;
            rw[i] = (kk < K)
                ? (WTR ? W[(size_t)(n0 + nn) * ldw + kk]
                       : W[(size_t)kk * ldw + (n0 + nn)]) : 0.f;
        }
        __syncthreads();   // previous users of Wt done
#pragma unroll
        for (int i = 0; i < LEL; ++i) {
            int e  = i * NTHREADS + threadIdx.x;
            int nn = e % NC, kk = e / NC;
            Wt[nn * WT_PITCH + kk] = f2bf(rw[i]);
        }
        __syncthreads();

        // ---- pipelined K loop: one barrier per slab ----
        for (int kt = 0; kt < ktiles; ++kt) {
            const int k0 = kt << 5;
            const unsigned short* cur = Wt + (kt & 1) * WT_BUFELEMS;
            const bool more = (kt + 1) < ktiles;

            if (more) {            // issue next slab's HBM loads NOW
                const int kb = k0 + 32;
#pragma unroll
                for (int i = 0; i < LEL; ++i) {
                    int e  = i * NTHREADS + threadIdx.x;
                    int nn = e % NC, kk = kb + e / NC;
                    rw[i] = (kk < K)
                        ? (WTR ? W[(size_t)(n0 + nn) * ldw + kk]
                               : W[(size_t)kk * ldw + (n0 + nn)]) : 0.f;
                }
                if (threadIdx.x == 0 && (kb + 32) < K) {   // prime L2, 2 ahead
                    const void* p = WTR
                        ? (const void*)&W[(size_t)n0 * ldw + (kb + 32)]
                        : (const void*)&W[(size_t)(kb + 32) * ldw + n0];
                    __builtin_prefetch(p, 0, 3);
                }
            }

            // ---- A fragment (16x32 bf16, contiguous k = 16h + s, h = lane/16) ----
            BF16x16 af;
            const unsigned short* arow =
                A + (size_t)(m0 + colin) * apitch + k0 + halfsel;
#pragma unroll
            for (int j = 0; j < 4; ++j)
                af.u[j] = *(const unsigned int*)(arow + 2 * j);
#pragma unroll
            for (int j = 0; j < 4; ++j)
                af.u[4 + j] = *(const unsigned int*)(arow + 8 + 2 * j);

            // ---- B fragments + WMMA burst (loads for kt+1 are in flight) --
#pragma unroll
            for (int nt = 0; nt < NTILES; ++nt) {
                BF16x16 bfm;
                const unsigned short* brow =
                    cur + (nt * 16 + colin) * WT_PITCH + ksel;
#pragma unroll
                for (int j = 0; j < 8; ++j)
                    bfm.u[j] = *(const unsigned int*)(brow + 2 * j);
                acc[nt] = __builtin_amdgcn_wmma_f32_16x16x32_bf16(
                    false, af.v, false, bfm.v, (short)0, acc[nt], false, false);
            }

            if (more) {            // convert + stage into the other buffer
                unsigned short* nxt = Wt + ((kt + 1) & 1) * WT_BUFELEMS;
#pragma unroll
                for (int i = 0; i < LEL; ++i) {
                    int e  = i * NTHREADS + threadIdx.x;
                    int nn = e % NC, kk = e / NC;
                    nxt[nn * WT_PITCH + kk] = f2bf(rw[i]);
                }
            }
            __syncthreads();       // slab kt fully consumed, kt+1 staged
        }

        // ------------------------------ epilogue ---------------------------
        if constexpr (MODE == 0) {            // relu -> LDS bf16
#pragma unroll
            for (int nt = 0; nt < NTILES; ++nt) {
                int col = n0 + nt * 16 + colin;
                float bb = bias ? bias[col] : 0.f;
#pragma unroll
                for (int j = 0; j < 8; ++j) {
                    int row = m0 + rofs + j;
                    float v = acc[nt][j] + bb;
                    O[(size_t)row * opitch + col] = f2bf(v > 0.f ? v : 0.f);
                }
            }
        } else if constexpr (MODE == 1) {     // sigmoid -> global out[b][f][n]
#pragma unroll
            for (int nt = 0; nt < NTILES; ++nt) {
                int col = n0 + nt * 16 + colin;
                float bb = bias[col];
#pragma unroll
                for (int j = 0; j < 8; ++j) {
                    int row = m0 + rofs + j;
                    float v = acc[nt][j] + bb;
                    gout[(((size_t)(b0 + row)) * FTS + f) * FTS + col] =
                        1.f / (1.f + __expf(-v));
                }
            }
        } else {                              // VQ scores + running argmin
#pragma unroll
            for (int nt = 0; nt < NTILES; ++nt) {
                int colc = nt * 16 + colin;
                float cn = cnorm[n0 + colc];
#pragma unroll
                for (int j = 0; j < 8; ++j) {
                    int row = m0 + rofs + j;
                    Sc[row * NC + colc] = cn - 2.f * acc[nt][j];
                }
            }
            __syncthreads();
            {   // 2 threads per row scan NC/2 columns each
                int t  = threadIdx.x;
                int rr = t >> 1, hh = t & 1;
                float mv = 3.4e38f; int mi2 = 0;
                int cbeg = hh * (NC / 2), cend = cbeg + NC / 2;
                for (int c = cbeg; c < cend; ++c) {
                    float s = Sc[rr * NC + c];
                    if (s < mv) { mv = s; mi2 = n0 + c; }
                }
                redv[t] = mv; redi[t] = mi2;
            }
            __syncthreads();
            if (threadIdx.x < MTILE) {
                int rr = threadIdx.x;
                float a0 = redv[2 * rr], a1 = redv[2 * rr + 1];
                float mv = a0; int mi2 = redi[2 * rr];
                if (a1 < mv) { mv = a1; mi2 = redi[2 * rr + 1]; }
                if (mv < minv[rr]) { minv[rr] = mv; mini[rr] = mi2; }
            }
            __syncthreads();
        }
    }
}

// ---------------------------------------------------------------------------
// Fused megakernel: one block = (feature f, half of the batch).
// ---------------------------------------------------------------------------
__global__ __launch_bounds__(NTHREADS, 1)
void vqvae_fused(const float* __restrict__ x,
                 const float* __restrict__ w1, const float* __restrict__ b1,
                 const float* __restrict__ w2, const float* __restrict__ b2,
                 const float* __restrict__ w3, const float* __restrict__ b3,
                 const float* __restrict__ cb,
                 const float* __restrict__ w4, const float* __restrict__ b4,
                 const float* __restrict__ w5, const float* __restrict__ b5,
                 const float* __restrict__ w6, const float* __restrict__ b6,
                 float* __restrict__ out)
{
    const int f  = blockIdx.x >> 1;
    const int b0 = (blockIdx.x & 1) * MTILE;

    // ---------------- LDS carve (297 KB of the 320 KB WGP LDS) -------------
    extern __shared__ unsigned char smem_raw[];
    unsigned short* Abuf = (unsigned short*)(smem_raw);            // 136 KB
    unsigned short* Bbuf = (unsigned short*)(smem_raw + 139264);   // 136 KB
    unsigned short* Wt   = (unsigned short*)(smem_raw + 278528);   // 10 KB x2
    float* cnorm = (float*)(smem_raw + 299008);                    // 2 KB
    float* redv  = (float*)(smem_raw + 301056);                    // 1 KB
    int*   redi  = (int*)  (smem_raw + 302080);                    // 1 KB
    float* minv  = (float*)(smem_raw + 303104);                    // 0.5 KB
    int*   mini  = (int*)  (smem_raw + 303616);                    // 0.5 KB
    float* Sc    = (float*)Abuf;   // 64 KB score buffer aliases dead Abuf

    // ---------------- stage x tile -> Abuf[128][264] bf16 (pad 255->256) ---
    for (int e = threadIdx.x; e < MTILE * 256; e += NTHREADS) {
        int r = e >> 8, i = e & 255;
        float v = (i < IN_DIM)
            ? x[(((size_t)(b0 + r)) * FTS + f) * IN_DIM + i] : 0.f;
        Abuf[r * (256 + APAD) + i] = f2bf(v);
    }
    // ---------------- codebook row norms -----------------------------------
    for (int n = threadIdx.x; n < EMB; n += NTHREADS) {
        float s = 0.f;
        for (int c = 0; c < DIMQ; ++c) { float v = cb[n * DIMQ + c]; s += v * v; }
        cnorm[n] = s;
    }
    if (threadIdx.x < MTILE) { minv[threadIdx.x] = 3.4e38f; mini[threadIdx.x] = 0; }
    __syncthreads();

    // ------------------------------ encoder --------------------------------
    fat_gemm<128, 0, 0>(w1 + (size_t)f * IN_DIM * U0, IN_DIM, U0, U0,
                        Abuf, 256 + APAD, Bbuf, U0 + APAD, b1 + (size_t)f * U0,
                        nullptr, f, b0, Wt, Sc, cnorm, redv, redi, minv, mini);
    __syncthreads();
    fat_gemm<128, 0, 0>(w2 + (size_t)f * U0 * U1, U0, U1, U1,
                        Bbuf, U0 + APAD, Abuf, U1 + APAD, b2 + (size_t)f * U1,
                        nullptr, f, b0, Wt, Sc, cnorm, redv, redi, minv, mini);
    __syncthreads();
    fat_gemm<64, 0, 0>(w3 + (size_t)f * U1 * DIMQ, U1, DIMQ, DIMQ,
                       Abuf, U1 + APAD, Bbuf, DIMQ + APAD, b3 + (size_t)f * DIMQ,
                       nullptr, f, b0, Wt, Sc, cnorm, redv, redi, minv, mini);
    __syncthreads();

    // --------------- VQ: scores = |c|^2 - 2 * h3 . c  (GEMM vs cb^T) -------
    fat_gemm<128, 2, 1>(cb, DIMQ, EMB, DIMQ,
                        Bbuf, DIMQ + APAD, nullptr, 0, nullptr,
                        nullptr, f, b0, Wt, Sc, cnorm, redv, redi, minv, mini);
    __syncthreads();
    // gather q = codebook[argmin] -> Abuf[128][72] bf16
    for (int e = threadIdx.x; e < MTILE * DIMQ; e += NTHREADS) {
        int r = e >> 6, c = e & 63;
        Abuf[r * (DIMQ + APAD) + c] = f2bf(cb[(size_t)mini[r] * DIMQ + c]);
    }
    __syncthreads();

    // ------------------------------ decoder --------------------------------
    fat_gemm<128, 0, 0>(w4 + (size_t)f * DIMQ * U1, DIMQ, U1, U1,
                        Abuf, DIMQ + APAD, Bbuf, U1 + APAD, b4 + (size_t)f * U1,
                        nullptr, f, b0, Wt, Sc, cnorm, redv, redi, minv, mini);
    __syncthreads();
    fat_gemm<128, 0, 0>(w5 + (size_t)f * U1 * U0, U1, U0, U0,
                        Bbuf, U1 + APAD, Abuf, U0 + APAD, b5 + (size_t)f * U0,
                        nullptr, f, b0, Wt, Sc, cnorm, redv, redi, minv, mini);
    __syncthreads();
    fat_gemm<128, 1, 0>(w6 + (size_t)f * U0 * FTS, U0, FTS, FTS,
                        Abuf, U0 + APAD, nullptr, 0, b6 + (size_t)f * FTS,
                        out, f, b0, Wt, Sc, cnorm, redv, redi, minv, mini);
}

extern "C" void kernel_launch(void* const* d_in, const int* in_sizes, int n_in,
                              void* d_out, int out_size, void* d_ws, size_t ws_size,
                              hipStream_t stream) {
    (void)in_sizes; (void)n_in; (void)out_size; (void)d_ws; (void)ws_size;
    const float* x  = (const float*)d_in[0];
    const float* w1 = (const float*)d_in[1];
    const float* b1 = (const float*)d_in[2];
    const float* w2 = (const float*)d_in[3];
    const float* b2 = (const float*)d_in[4];
    const float* w3 = (const float*)d_in[5];
    const float* b3 = (const float*)d_in[6];
    const float* cb = (const float*)d_in[7];
    const float* w4 = (const float*)d_in[8];
    const float* b4 = (const float*)d_in[9];
    const float* w5 = (const float*)d_in[10];
    const float* b5 = (const float*)d_in[11];
    const float* w6 = (const float*)d_in[12];
    const float* b6 = (const float*)d_in[13];
    float* out = (float*)d_out;

    const size_t shmem = 304128;   // 297 KB carve (see kernel)
    dim3 grid(FTS * (BATCH / MTILE));   // 512 blocks: (feature, batch-half)
    dim3 block(NTHREADS);
    vqvae_fused<<<grid, block, shmem, stream>>>(
        x, w1, b1, w2, b2, w3, b3, cb, w4, b4, w5, b5, w6, b6, out);
}
